// CausalMultiheadSelfAttention_51591147159702
// MI455X (gfx1250) — matrix operand order repaired and checked by reading, not compiled, not run on hardware
//
#include <hip/hip_runtime.h>
#include <hip/hip_bf16.h>

// Causal multi-head self-attention, WMMA-f16 pipeline for gfx1250 (wave32).
// Pipeline: transpose+cvt weights -> cvt x -> 3x tiled GEMM (Q,K,V; V stored
// head-transposed) -> flash-attention per 16-query tile -> output GEMM (f32).
// GEMMs stage the shared B-tile into LDS with async global->LDS copies
// (ASYNCcnt), double-buffered across the K loop.

#define BATCH     2
#define NUM_HEADS 16
#define SEQ       2048
#define D_MODEL   1024
#define DK        64
#define MROWS     (BATCH * SEQ)   // 4096

typedef __attribute__((ext_vector_type(16))) _Float16 v16h;
typedef __attribute__((ext_vector_type(8)))  _Float16 v8h;
typedef __attribute__((ext_vector_type(8)))  float    v8f;
typedef __attribute__((ext_vector_type(4)))  int      v4i;

typedef __attribute__((address_space(1))) v4i gas_v4i;   // global
typedef __attribute__((address_space(3))) v4i las_v4i;   // LDS

__device__ __forceinline__ v16h cat16(v8h lo, v8h hi) {
    v16h r;
#pragma unroll
    for (int i = 0; i < 8; ++i) { r[i] = lo[i]; r[i + 8] = hi[i]; }
    return r;
}

// A-operand fragment (16x32 f16): lane half h holds k in [8h,8h+8) U [16+8h,16+8h+8)
__device__ __forceinline__ v16h load_a(const _Float16* base, int ld, int row, int kb, int half) {
    const _Float16* p = base + (size_t)row * ld + kb + 8 * half;
    return cat16(*(const v8h*)p, *(const v8h*)(p + 16));
}

// B-operand fragment (32x16 f16, column 'row' of B == row of Bt): lane half h holds k in [8h,8h+8) U [16+8h,16+8h+8)
__device__ __forceinline__ v16h load_b(const _Float16* base, int ld, int row, int kb, int half) {
    const _Float16* p = base + (size_t)row * ld + kb + 8 * half;
    return cat16(*(const v8h*)p, *(const v8h*)(p + 16));
}

__device__ __forceinline__ v8f wmma32(v16h a, v16h b, v8f c) {
    return __builtin_amdgcn_wmma_f32_16x16x32_f16(
        /*neg_a=*/false, a, /*neg_b=*/false, b,
        /*c_mod=*/(short)0, c, /*reuse_a=*/false, /*reuse_b=*/false);
}

// ---- async global -> LDS (16B per call), ASYNCcnt-tracked -------------------
// lds_off: byte offset into the kernel's LDS allocation (AS(3) pointers are
// 32-bit LDS offsets); g: global source address.
__device__ __forceinline__ void async_copy16(unsigned lds_off, const _Float16* g) {
#if __has_builtin(__builtin_amdgcn_global_load_async_to_lds_b128)
    __builtin_amdgcn_global_load_async_to_lds_b128(
        (gas_v4i*)(uintptr_t)g, (las_v4i*)(uintptr_t)lds_off, 0, 0);
#else
    asm volatile("global_load_async_to_lds_b128 %0, %1, off"
                 :: "v"(lds_off), "v"(g) : "memory");
#endif
}

__device__ __forceinline__ void wait_async_0() {
#if __has_builtin(__builtin_amdgcn_s_wait_asynccnt)
    __builtin_amdgcn_s_wait_asynccnt(0);
#else
    asm volatile("s_wait_asynccnt 0" ::: "memory");
#endif
}
__device__ __forceinline__ void wait_async_2() {
#if __has_builtin(__builtin_amdgcn_s_wait_asynccnt)
    __builtin_amdgcn_s_wait_asynccnt(2);
#else
    asm volatile("s_wait_asynccnt 2" ::: "memory");
#endif
}

// ---- prep kernels -----------------------------------------------------------

// Wt[n][k] = (f16) W[k][n]   (so B-operand columns become contiguous rows)
__global__ void k_twt(const float* __restrict__ W, _Float16* __restrict__ Wt) {
    int idx = blockIdx.x * 256 + threadIdx.x;       // idx = n*1024 + k
    int n = idx >> 10;
    int k = idx & (D_MODEL - 1);
    Wt[idx] = (_Float16)W[(size_t)k * D_MODEL + n];
}

__global__ void k_cvt(const float* __restrict__ X, _Float16* __restrict__ Xh, int n) {
    int idx = blockIdx.x * 256 + threadIdx.x;
    if (idx < n) Xh[idx] = (_Float16)X[idx];
}

// ---- tiled GEMM: C[4096x1024] = A[4096x1024] * Bt^T -------------------------
// Block = 4 waves; block tile 128(M) x 64(N); per-wave 32x64.
// Bt tile (64 rows x 32 k = 4KB) async-staged to LDS, double buffered.
// MODE 0: store f16 row-major; MODE 1: store f16 as Vt[b][h][d][s]; MODE 2: f32.
template <int MODE>
__global__ __launch_bounds__(128) void k_gemm(const _Float16* __restrict__ A,
                                              const _Float16* __restrict__ Bt,
                                              void* __restrict__ Cout) {
    __shared__ _Float16 Blds[2][64 * 32];

    const int tid  = (int)threadIdx.x;
    const int wave = tid >> 5;
    const int lane = tid & 31;
    const int half = lane >> 4;
    const int nl   = lane & 15;
    const int m0 = blockIdx.y * 128 + wave * 32;
    const int n0 = blockIdx.x * 64;

    // stage Bt[n0..n0+63][kb..kb+31] into Blds[buf]: 256 16B-chunks / 128 thr
    auto stage = [&](int buf, int kb) {
#pragma unroll
        for (int i = 0; i < 2; ++i) {
            int c  = tid * 2 + i;          // 0..255
            int n  = c >> 2;               // row 0..63
            int ko = (c & 3) * 8;          // k offset in halves
            unsigned off = (unsigned)((uintptr_t)&Blds[buf][n * 32 + ko] -
                                      (uintptr_t)&Blds[0][0]);
            async_copy16(off, Bt + (size_t)(n0 + n) * D_MODEL + kb + ko);
        }
    };

    v8f acc[2][4] = {};
    stage(0, 0);

    for (int kb = 0; kb < D_MODEL; kb += 32) {
        const int buf = (kb >> 5) & 1;
        if (kb + 32 < D_MODEL) {
            stage(buf ^ 1, kb + 32);   // prefetch next tile
            wait_async_2();            // current tile's 2 copies done
        } else {
            wait_async_0();
        }
        __syncthreads();               // all waves' copies visible

        v16h a0 = load_a(A, D_MODEL, m0 + nl,      kb, half);
        v16h a1 = load_a(A, D_MODEL, m0 + 16 + nl, kb, half);
#pragma unroll
        for (int c = 0; c < 4; ++c) {
            v16h bf = load_b(&Blds[buf][0], 32, c * 16 + nl, 0, half);  // ds_load_b128 x2
            acc[0][c] = wmma32(a0, bf, acc[0][c]);
            acc[1][c] = wmma32(a1, bf, acc[1][c]);
        }
        __syncthreads();               // tile consumed before next overwrite
    }

#pragma unroll
    for (int mt = 0; mt < 2; ++mt)
#pragma unroll
        for (int c = 0; c < 4; ++c)
#pragma unroll
            for (int r = 0; r < 8; ++r) {
                int gm = m0 + mt * 16 + half * 8 + r;
                int gn = n0 + c * 16 + nl;
                float v = acc[mt][c][r];
                if (MODE == 2) {
                    ((float*)Cout)[(size_t)gm * D_MODEL + gn] = v;
                } else if (MODE == 0) {
                    ((_Float16*)Cout)[(size_t)gm * D_MODEL + gn] = (_Float16)v;
                } else {
                    int b = gm >> 11, s = gm & (SEQ - 1);   // gm = b*SEQ + s
                    int hh = gn >> 6, d = gn & (DK - 1);    // gn = hh*DK + d
                    ((_Float16*)Cout)[((size_t)(b * NUM_HEADS + hh) * DK + d) * SEQ + s] =
                        (_Float16)v;
                }
            }
}

// ---- flash attention: one wave handles 16 query rows of one (b,h) -----------
__global__ __launch_bounds__(32) void k_attn(const _Float16* __restrict__ Q,
                                             const _Float16* __restrict__ Kx,
                                             const _Float16* __restrict__ Vt,
                                             _Float16* __restrict__ Oh) {
    __shared__ _Float16 Plds[16][32];

    const int lane = (int)(threadIdx.x & 31u);
    const int half = lane >> 4;
    const int nl   = lane & 15;
    const int q0 = blockIdx.x * 16;
    const int bh = blockIdx.y;          // b*H + h
    const int b  = bh >> 4;
    const int h  = bh & 15;

    const _Float16* Qb = Q  + (size_t)b * SEQ * D_MODEL + h * DK;   // [S][D], head slice
    const _Float16* Kb = Kx + (size_t)b * SEQ * D_MODEL + h * DK;
    const _Float16* Vb = Vt + (size_t)bh * DK * SEQ;                // [DK][S]

    // Q fragments (A operand), dk split into two k-steps
    v16h qf0 = load_a(Qb, D_MODEL, q0 + nl, 0,  half);
    v16h qf1 = load_a(Qb, D_MODEL, q0 + nl, 32, half);

    float mrow[8], lrow[8];
    v8f oacc[4] = {};
#pragma unroll
    for (int r = 0; r < 8; ++r) { mrow[r] = -1e30f; lrow[r] = 0.0f; }

    const int jend = q0 + 16;                       // causal: keys < q0+16
    for (int j = 0; j < jend; j += 32) {
        // scores: two 16x16 tiles (cols j..j+15, j+16..j+31)
        v8f s0 = {}, s1 = {};
        v16h kf;
        kf = load_b(Kb, D_MODEL, j + nl,      0,  half); s0 = wmma32(qf0, kf, s0);
        kf = load_b(Kb, D_MODEL, j + nl,      32, half); s0 = wmma32(qf1, kf, s0);
        kf = load_b(Kb, D_MODEL, j + 16 + nl, 0,  half); s1 = wmma32(qf0, kf, s1);
        kf = load_b(Kb, D_MODEL, j + 16 + nl, 32, half); s1 = wmma32(qf1, kf, s1);

#pragma unroll
        for (int r = 0; r < 8; ++r) {
            const int qrow = q0 + 8 * half + r;
            float a = s0[r] * 0.125f;               // 1/sqrt(64)
            float c = s1[r] * 0.125f;
            if (j + nl > qrow)      a = -1e30f;     // causal mask
            if (j + 16 + nl > qrow) c = -1e30f;

            // row max across the 16 lanes of this half
            float v = fmaxf(a, c);
#pragma unroll
            for (int msk = 1; msk < 16; msk <<= 1) v = fmaxf(v, __shfl_xor(v, msk, 16));

            float mnew = fmaxf(mrow[r], v);
            float corr = __expf(mrow[r] - mnew);
            mrow[r] = mnew;
            float p0 = __expf(a - mnew);
            float p1 = __expf(c - mnew);
            float t = p0 + p1;
#pragma unroll
            for (int msk = 1; msk < 16; msk <<= 1) t += __shfl_xor(t, msk, 16);
            lrow[r] = lrow[r] * corr + t;
#pragma unroll
            for (int cc = 0; cc < 4; ++cc) oacc[cc][r] *= corr;

            // spill P (D-layout) to LDS so it can be re-read in A-layout
            Plds[8 * half + r][nl]      = (_Float16)p0;
            Plds[8 * half + r][16 + nl] = (_Float16)p1;
        }
        __builtin_amdgcn_wave_barrier();

        v16h pf = load_a(&Plds[0][0], 32, nl, 0, half);   // ds_load_b128 x2
#pragma unroll
        for (int c = 0; c < 4; ++c) {
            v16h vf = load_b(Vb, SEQ, c * 16 + nl, j, half);
            oacc[c] = wmma32(pf, vf, oacc[c]);
        }
        __builtin_amdgcn_wave_barrier();                  // Plds reused next iter
    }

    // normalize and store head output (f16, [M][D] layout, head column slice)
#pragma unroll
    for (int r = 0; r < 8; ++r) {
        float inv = 1.0f / lrow[r];
        size_t gm = (size_t)b * SEQ + q0 + 8 * half + r;
#pragma unroll
        for (int c = 0; c < 4; ++c)
            Oh[gm * D_MODEL + h * DK + c * 16 + nl] = (_Float16)(oacc[c][r] * inv);
    }
}

// ---- launch -----------------------------------------------------------------
extern "C" void kernel_launch(void* const* d_in, const int* in_sizes, int n_in,
                              void* d_out, int out_size, void* d_ws, size_t ws_size,
                              hipStream_t stream) {
    const float* x  = (const float*)d_in[0];
    const float* Wq = (const float*)d_in[1];
    const float* Wk = (const float*)d_in[2];
    const float* Wv = (const float*)d_in[3];
    const float* Wo = (const float*)d_in[4];
    float* out = (float*)d_out;

    const size_t WW = (size_t)D_MODEL * D_MODEL;   // 1M elems
    const size_t MD = (size_t)MROWS * D_MODEL;     // 4M elems
    const size_t need = (4 * WW + 5 * MD) * sizeof(_Float16);  // 48 MB
    if (ws_size < need) return;

    _Float16* ws  = (_Float16*)d_ws;
    _Float16* Wqt = ws;
    _Float16* Wkt = Wqt + WW;
    _Float16* Wvt = Wkt + WW;
    _Float16* Wot = Wvt + WW;
    _Float16* Xh  = Wot + WW;
    _Float16* Qh  = Xh + MD;
    _Float16* Kh  = Qh + MD;
    _Float16* Vt  = Kh + MD;     // [B][H][DK][SEQ]
    _Float16* Ah  = Vt + MD;

    k_twt<<<(int)(WW / 256), 256, 0, stream>>>(Wq, Wqt);
    k_twt<<<(int)(WW / 256), 256, 0, stream>>>(Wk, Wkt);
    k_twt<<<(int)(WW / 256), 256, 0, stream>>>(Wv, Wvt);
    k_twt<<<(int)(WW / 256), 256, 0, stream>>>(Wo, Wot);
    k_cvt<<<(int)(MD / 256), 256, 0, stream>>>(x, Xh, (int)MD);

    dim3 gg(D_MODEL / 64, MROWS / 128);
    k_gemm<0><<<gg, 128, 0, stream>>>(Xh, Wqt, (void*)Qh);
    k_gemm<0><<<gg, 128, 0, stream>>>(Xh, Wkt, (void*)Kh);
    k_gemm<1><<<gg, 128, 0, stream>>>(Xh, Wvt, (void*)Vt);

    dim3 ga(SEQ / 16, BATCH * NUM_HEADS);
    k_attn<<<ga, 32, 0, stream>>>(Qh, Kh, Vt, Ah);

    k_gemm<2><<<gg, 128, 0, stream>>>(Ah, Wot, (void*)out);
}
